// selfattention_36739150250382
// MI455X (gfx1250) — matrix operand order repaired and checked by reading, not compiled, not run on hardware
//
#include <hip/hip_runtime.h>
#include <math.h>
#include <stdint.h>

#define Bn 4
#define Hn 16
#define Sn 2048
#define Dn 128
#define NELEM (Bn * Hn * Sn * Dn)   // 16,777,216 elements per tensor
#define KT 64                        // keys per k-loop iteration

typedef __attribute__((ext_vector_type(16))) __bf16 v16bf;
typedef __attribute__((ext_vector_type(8)))  __bf16 v8bf;
typedef __attribute__((ext_vector_type(8)))  float  v8f;
typedef __attribute__((ext_vector_type(4)))  unsigned int u32x4;
typedef __attribute__((ext_vector_type(8)))  int i32x8;
typedef __attribute__((ext_vector_type(4)))  int i32x4;

union Frag16 { v16bf v; v8bf h[2]; };

#if defined(__has_builtin)
#if __has_builtin(__builtin_amdgcn_tensor_load_to_lds) && \
    __has_builtin(__builtin_amdgcn_s_wait_tensorcnt)
#define HAVE_TDM 1
#endif
#endif

// ---------------------------------------------------------------------------
// Prep: fp32 -> bf16.  Q pre-scaled by 1/sqrt(D); V transposed to [bh][d][s].
// ---------------------------------------------------------------------------
__global__ __launch_bounds__(256) void fa_prep(
    const float* __restrict__ Q, const float* __restrict__ K,
    const float* __restrict__ V, __bf16* __restrict__ Qb,
    __bf16* __restrict__ Kb, __bf16* __restrict__ Vt) {
  const float scale = 0.08838834764831845f;  // 1/sqrt(128)
  long long i = (long long)blockIdx.x * blockDim.x + threadIdx.x;
  if (i >= (long long)NELEM) return;
  Qb[i] = (__bf16)(Q[i] * scale);
  Kb[i] = (__bf16)K[i];
  int d        = (int)(i & (Dn - 1));
  long long sd = i >> 7;
  int s        = (int)(sd & (Sn - 1));
  long long bh = sd >> 11;
  Vt[(bh * Dn + d) * Sn + s] = (__bf16)V[i];
}

#ifdef HAVE_TDM
// 2D TDM load: tile (tile1 rows x tile0 elems) of a bf16 tensor whose row
// stride is stride0 elements, into LDS at byte offset lds_off (row-major,
// packed).  D# layout per CDNA5 ISA ch.8.
__device__ __forceinline__ void tdm_load_2d(uint32_t lds_off, const void* gaddr,
                                            uint32_t tdim0, uint32_t tdim1,
                                            uint64_t stride0, uint32_t tile0,
                                            uint32_t tile1) {
  uint64_t ga = (uint64_t)(uintptr_t)gaddr;
  u32x4 g0;
  g0.x = 1u;                                   // count=1 valid descriptor
  g0.y = lds_off;                              // lds_addr (bytes)
  g0.z = (uint32_t)ga;                         // global_addr[31:0]
  g0.w = (uint32_t)((ga >> 32) & 0x01FFFFFFu)  // global_addr[56:32]
         | (2u << 30);                         // type = 2 ("image")
  i32x8 g1;
  g1[0] = (int)(1u << 16);                                 // data_size = 2B
  g1[1] = (int)((tdim0 & 0xFFFFu) << 16);                  // tensor_dim0 lo
  g1[2] = (int)(((tdim0 >> 16) & 0xFFFFu) |
                ((tdim1 & 0xFFFFu) << 16));                // dim0 hi | dim1 lo
  g1[3] = (int)(((tdim1 >> 16) & 0xFFFFu) | (tile0 << 16)); // dim1 hi | tile_dim0
  g1[4] = (int)(tile1 & 0xFFFFu);                          // tile_dim1, tile_dim2=0
  g1[5] = (int)(uint32_t)stride0;                          // dim0_stride lo
  g1[6] = (int)(uint32_t)((stride0 >> 32) & 0xFFFFu);      // dim0_stride hi
  g1[7] = 0;
  i32x4 z4 = (i32x4){0, 0, 0, 0};
#if __clang_major__ >= 23
  i32x8 z8 = (i32x8){0, 0, 0, 0, 0, 0, 0, 0};
  __builtin_amdgcn_tensor_load_to_lds(g0, g1, z4, z4, z8, 0);
#else
  __builtin_amdgcn_tensor_load_to_lds(g0, g1, z4, z4, 0);
#endif
}
#endif

// 16-lane xor-reduction via ds_swizzle (group-of-32, and=0x1F, xor=m)
#define SWZ(x, imm) \
  __int_as_float(__builtin_amdgcn_ds_swizzle(__float_as_int(x), imm))
__device__ __forceinline__ float rmax16(float x) {
  x = fmaxf(x, SWZ(x, 0x041F));
  x = fmaxf(x, SWZ(x, 0x081F));
  x = fmaxf(x, SWZ(x, 0x101F));
  x = fmaxf(x, SWZ(x, 0x201F));
  return x;
}
__device__ __forceinline__ float rsum16(float x) {
  x += SWZ(x, 0x041F);
  x += SWZ(x, 0x081F);
  x += SWZ(x, 0x101F);
  x += SWZ(x, 0x201F);
  return x;
}

// ---------------------------------------------------------------------------
// Flash attention forward: 8 wave32 per block, 16 query rows per wave,
// K/V tiles (64 keys) staged in LDS by the Tensor Data Mover, double-buffered,
// with software-pipelined B-fragment loads under the WMMA chains.
// ---------------------------------------------------------------------------
__global__ __launch_bounds__(256) void fa_fwd(
    const __bf16* __restrict__ Qb, const __bf16* __restrict__ Kb,
    const __bf16* __restrict__ Vt, const int* __restrict__ maskp,
    float* __restrict__ Out) {
  __shared__ __align__(16) __bf16 bufK[2][KT * Dn];   // [keys][d]
  __shared__ __align__(16) __bf16 bufV[2][Dn * KT];   // [d][keys]
  __shared__ __align__(16) __bf16 ldsP[8][16 * KT];   // per-wave P tile

  const int tid  = threadIdx.x;
  const int wave = tid >> 5;
  const int lane = tid & 31;
  const int hl   = lane >> 4;
  const int lq   = lane & 15;
  const int bh   = blockIdx.y;
  const int qtile = blockIdx.x * 128 + wave * 16;
  const bool causal = (*maskp) != 0;

  const __bf16* Qh = Qb + (long long)bh * Sn * Dn;
  const __bf16* Kh = Kb + (long long)bh * Sn * Dn;
  const __bf16* Vh = Vt + (long long)bh * Dn * Sn;

  // ---- Q A-fragments ----
  Frag16 qf[4];
  {
    const __bf16* qrow = Qh + (long long)(qtile + lq) * Dn;
#pragma unroll
    for (int f = 0; f < 4; ++f) {
      const int k0 = 32 * f + 16 * hl;
      qf[f].h[0] = *(const v8bf*)(qrow + k0);
      qf[f].h[1] = *(const v8bf*)(qrow + k0 + 8);
    }
  }

  v8f acc[8];
#pragma unroll
  for (int i = 0; i < 8; ++i) acc[i] = (v8f){};
  float mrow[8], lrow[8];
#pragma unroll
  for (int r = 0; r < 8; ++r) { mrow[r] = -INFINITY; lrow[r] = 0.f; }

  __bf16* myP = &ldsP[wave][0];

  // block-uniform causal bound (multiple of KT)
  const int kend = causal ? (blockIdx.x + 1) * 128 : Sn;

#ifdef HAVE_TDM
  if (wave == 0) {  // prologue: tile 0 -> buffer 0
    tdm_load_2d((uint32_t)(uintptr_t)&bufK[0][0], Kh, Dn, Sn, Dn, Dn, KT);
    tdm_load_2d((uint32_t)(uintptr_t)&bufV[0][0], Vh, Sn, Dn, Sn, KT, Dn);
  }
#endif

  int it = 0;
  for (int c0 = 0; c0 < kend; c0 += KT, ++it) {
    const int b = it & 1;
    const __bf16* bK = &bufK[b][0];
    const __bf16* bV = &bufV[b][0];

#ifdef HAVE_TDM
    if (wave == 0) __builtin_amdgcn_s_wait_tensorcnt(0);
    __syncthreads();                     // buffer b ready for everyone
    if (wave == 0 && c0 + KT < kend) {   // prefetch next tile -> buffer b^1
      tdm_load_2d((uint32_t)(uintptr_t)&bufK[b ^ 1][0],
                  Kh + (long long)(c0 + KT) * Dn, Dn, Sn, Dn, Dn, KT);
      tdm_load_2d((uint32_t)(uintptr_t)&bufV[b ^ 1][0],
                  Vh + (c0 + KT), Sn, Dn, Sn, KT, Dn);
    }
#else
    __syncthreads();                     // previous compute done
    for (int i = tid; i < (KT * Dn) / 8; i += 256)
      ((uint4*)&bufK[b][0])[i] = *(const uint4*)(Kh + (long long)c0 * Dn + 8 * i);
    for (int i = tid; i < (Dn * KT) / 8; i += 256) {
      int d = i >> 3, j = i & 7;
      ((uint4*)&bufV[b][0])[i] = *(const uint4*)(Vh + (long long)d * Sn + c0 + 8 * j);
    }
    __syncthreads();
#endif

    // ---- scores: four 16x16 tiles over KT=64 keys (pipelined frag loads) ----
    v8f sc[4];
#pragma unroll
    for (int t = 0; t < 4; ++t) sc[t] = (v8f){};
    {
      Frag16 kb[2];
      kb[0].v = *(const v16bf*)(bK + lq * Dn + 16 * hl);  // (t=0,f=0)
#pragma unroll
      for (int i = 0; i < 16; ++i) {
        if (i + 1 < 16) {
          const int tn = (i + 1) >> 2, fn = (i + 1) & 3;
          kb[(i + 1) & 1].v =
              *(const v16bf*)(bK + (16 * tn + lq) * Dn + 32 * fn + 16 * hl);
        }
        const int t = i >> 2, f = i & 3;
        sc[t] = __builtin_amdgcn_wmma_f32_16x16x32_bf16(
            false, qf[f].v, false, kb[i & 1].v, (short)0, sc[t], false, false);
      }
    }

    // ---- causal mask ----
    if (causal) {
#pragma unroll
      for (int t = 0; t < 4; ++t) {
        const int col = c0 + 16 * t + lq;
#pragma unroll
        for (int r = 0; r < 8; ++r)
          if (col > qtile + 8 * hl + r) sc[t][r] = -INFINITY;
      }
    }

    // ---- online softmax ----
#pragma unroll
    for (int r = 0; r < 8; ++r) {
      float tmax = fmaxf(fmaxf(sc[0][r], sc[1][r]), fmaxf(sc[2][r], sc[3][r]));
      tmax = rmax16(tmax);
      const float mnew  = fmaxf(mrow[r], tmax);
      const float alpha = __expf(mrow[r] - mnew);
      mrow[r] = mnew;
      float rs = 0.f;
#pragma unroll
      for (int t = 0; t < 4; ++t) {
        const float p = __expf(sc[t][r] - mnew);
        sc[t][r] = p;
        rs += p;
      }
      rs = rsum16(rs);
      lrow[r] = lrow[r] * alpha + rs;
#pragma unroll
      for (int dt = 0; dt < 8; ++dt) acc[dt][r] *= alpha;
    }

    // ---- P (C-layout) -> LDS row-major bf16 [16][KT] -> A-fragments ----
#pragma unroll
    for (int t = 0; t < 4; ++t)
#pragma unroll
      for (int r = 0; r < 8; ++r)
        myP[(8 * hl + r) * KT + 16 * t + lq] = (__bf16)sc[t][r];

    Frag16 pa[2];  // same-wave DS ordering: no barrier needed
#pragma unroll
    for (int c = 0; c < 2; ++c) {
      pa[c].h[0] = *(const v8bf*)(myP + lq * KT + 32 * c + 16 * hl);
      pa[c].h[1] = *(const v8bf*)(myP + lq * KT + 32 * c + 16 * hl + 8);
    }

    // ---- P(16x64) x V(64x128) (pipelined frag loads) ----
    {
      Frag16 vb[2];
      vb[0].v = *(const v16bf*)(bV + lq * KT + 16 * hl);  // (dt=0,c=0)
#pragma unroll
      for (int i = 0; i < 16; ++i) {
        if (i + 1 < 16) {
          const int dn = (i + 1) >> 1, cn = (i + 1) & 1;
          vb[(i + 1) & 1].v =
              *(const v16bf*)(bV + (16 * dn + lq) * KT + 32 * cn + 16 * hl);
        }
        const int dt = i >> 1, c = i & 1;
        acc[dt] = __builtin_amdgcn_wmma_f32_16x16x32_bf16(
            false, pa[c].v, false, vb[i & 1].v, (short)0, acc[dt], false, false);
      }
    }

    __syncthreads();  // everyone done with buffer b before it is refilled
  }

  // ---- epilogue ----
  float* orow = Out + (long long)bh * Sn * Dn;
#pragma unroll
  for (int r = 0; r < 8; ++r) {
    const float inv = 1.f / lrow[r];
    const int row = qtile + 8 * hl + r;
#pragma unroll
    for (int dt = 0; dt < 8; ++dt)
      orow[(long long)row * Dn + 16 * dt + lq] = acc[dt][r] * inv;
  }
}

// ---------------------------------------------------------------------------
extern "C" void kernel_launch(void* const* d_in, const int* in_sizes, int n_in,
                              void* d_out, int out_size, void* d_ws,
                              size_t ws_size, hipStream_t stream) {
  const float* Q = (const float*)d_in[0];
  const float* K = (const float*)d_in[1];
  const float* V = (const float*)d_in[2];
  const int* mask = (const int*)d_in[3];

  __bf16* Qb = (__bf16*)d_ws;
  __bf16* Kb = Qb + (size_t)NELEM;
  __bf16* Vt = Kb + (size_t)NELEM;

  fa_prep<<<NELEM / 256, 256, 0, stream>>>(Q, K, V, Qb, Kb, Vt);

  dim3 grid(Sn / 128, Bn * Hn);
  fa_fwd<<<grid, 256, 0, stream>>>(Qb, Kb, Vt, mask, (float*)d_out);
}
